// CGLayer_83786222010450
// MI455X (gfx1250) — compile-verified, matrix operand order checked by reading, not run on hardware
//
#include <hip/hip_runtime.h>

typedef __attribute__((ext_vector_type(16))) _Float16 v16h;
typedef __attribute__((ext_vector_type(8)))  float    v8f;

#define B_   4
#define N_   16384
#define M_   1024
#define C_   256
#define K_   32
#define BM_  (B_ * M_)          // 4096
#define ROWS (BM_ * K_)         // 131072
#define KC1  288                // 259 padded to 9*32
#define OC   512

// ---------------------------------------------------------------- weights
__global__ void cvt_w1_kernel(const float* __restrict__ w, _Float16* __restrict__ wh) {
    int o = blockIdx.x;          // 512
    int c = threadIdx.x;         // 288
    float v = (c < 259) ? w[o * 259 + c] : 0.0f;
    wh[o * KC1 + c] = (_Float16)v;
}

__global__ void cvt_w_kernel(const float* __restrict__ w, _Float16* __restrict__ wh, int n) {
    int t = blockIdx.x * blockDim.x + threadIdx.x;
    if (t < n) wh[t] = (_Float16)w[t];
}

// ---------------------------------------------------------------- shift MLP
// h1pre[o][bm] = sum_c sw1[o][c] * feat[b][c][m]
__global__ void shift1_kernel(const float* __restrict__ sw1, const float* __restrict__ feat,
                              float* __restrict__ h1pre) {
    int t  = blockIdx.x * blockDim.x + threadIdx.x;   // 128*4096
    int o  = t >> 12;
    int bm = t & (BM_ - 1);
    int b  = bm >> 10;
    int m  = bm & (M_ - 1);
    float s = 0.f;
    for (int c = 0; c < C_; ++c)
        s += sw1[o * C_ + c] * feat[((size_t)b * C_ + c) * M_ + m];
    h1pre[(size_t)o * BM_ + bm] = s;
}

// per-channel mean/var over n values, layout x[o][n]; emits scale/bias pairs
__global__ void stats_chan_kernel(const float* __restrict__ x, const float* __restrict__ g,
                                  const float* __restrict__ bt, float* __restrict__ sb, int n) {
    __shared__ float ss[256], sq[256];
    int o = blockIdx.x;
    float s = 0.f, q = 0.f;
    for (int i = threadIdx.x; i < n; i += 256) {
        float v = x[(size_t)o * n + i];
        s += v; q += v * v;
    }
    ss[threadIdx.x] = s; sq[threadIdx.x] = q;
    __syncthreads();
    for (int st = 128; st > 0; st >>= 1) {
        if ((int)threadIdx.x < st) { ss[threadIdx.x] += ss[threadIdx.x + st]; sq[threadIdx.x] += sq[threadIdx.x + st]; }
        __syncthreads();
    }
    if (threadIdx.x == 0) {
        float mean = ss[0] / n;
        float var  = sq[0] / n - mean * mean;
        float sc   = g[o] * rsqrtf(var + 1e-5f);
        sb[2 * o]     = sc;
        sb[2 * o + 1] = bt[o] - mean * sc;
    }
}

__global__ void shift2_kernel(const float* __restrict__ sw2, const float* __restrict__ h1pre,
                              const float* __restrict__ sb1, float* __restrict__ h2pre) {
    int t  = blockIdx.x * blockDim.x + threadIdx.x;   // 3*4096
    int o  = t >> 12;
    int bm = t & (BM_ - 1);
    float s = 0.f;
    for (int c = 0; c < 128; ++c) {
        float v = h1pre[(size_t)c * BM_ + bm];
        v = fmaxf(0.f, v * sb1[2 * c] + sb1[2 * c + 1]);
        s += sw2[o * 128 + c] * v;
    }
    h2pre[(size_t)o * BM_ + bm] = s;
}

__global__ void newxyz_kernel(const float* __restrict__ fxyz, const float* __restrict__ h2pre,
                              const float* __restrict__ sb2, float* __restrict__ nxyz) {
    int bm = blockIdx.x * blockDim.x + threadIdx.x;   // 4096
    for (int d = 0; d < 3; ++d) {
        float v = h2pre[(size_t)d * BM_ + bm];
        v = fmaxf(0.f, v * sb2[2 * d] + sb2[2 * d + 1]);
        nxyz[bm * 3 + d] = fxyz[bm * 3 + d] + v;
    }
}

// ---------------------------------------------------------------- ball query: one wave32 per query
__global__ void ballquery_kernel(const float* __restrict__ nxyz, const float* __restrict__ bxyz,
                                 int* __restrict__ idx) {
    int wave = (blockIdx.x * blockDim.x + threadIdx.x) >> 5;  // 4096 waves
    int lane = threadIdx.x & 31;
    int b = wave >> 10;
    float qx = nxyz[wave * 3 + 0];
    float qy = nxyz[wave * 3 + 1];
    float qz = nxyz[wave * 3 + 2];
    const float* P = bxyz + (size_t)b * N_ * 3;
    int* out = idx + (size_t)wave * K_;
    int cnt = 0, first = 0;
    for (int base = 0; base < N_; base += 32) {
        int j = base + lane;
        float dx = P[j * 3 + 0] - qx;
        float dy = P[j * 3 + 1] - qy;
        float dz = P[j * 3 + 2] - qz;
        bool pred = (dx * dx + dy * dy + dz * dz) < 1.0f;
        unsigned mask = (unsigned)__ballot(pred);
        if (cnt == 0 && mask) first = base + __ffs(mask) - 1;
        int pos = cnt + __popc(mask & ((1u << lane) - 1u));
        if (pred && pos < K_) out[pos] = j;
        cnt += __popc(mask);
        if (cnt >= K_) break;
    }
    if (cnt > K_) cnt = K_;
    if (lane >= cnt) out[lane] = first;   // cnt==0 -> first==0 (reference fill rule)
}

// ---------------------------------------------------------------- gather -> f16 A matrix [ROWS][288]
__global__ void gather_kernel(const float* __restrict__ bxyz, const float* __restrict__ bfeat,
                              const float* __restrict__ nxyz, const int* __restrict__ idx,
                              _Float16* __restrict__ g0) {
    int row = blockIdx.x;        // 131072
    int c   = threadIdx.x;       // 288
    int bm  = row >> 5;
    int k   = row & 31;
    int b   = bm >> 10;
    int j   = idx[(size_t)bm * K_ + k];
    float v;
    if (c < 3)        v = bxyz[((size_t)b * N_ + j) * 3 + c] - nxyz[bm * 3 + c];
    else if (c < 259) v = bfeat[((size_t)b * C_ + (c - 3)) * N_ + j];
    else              v = 0.f;
    g0[(size_t)row * KC1 + c] = (_Float16)v;
}

// ---------------------------------------------------------------- WMMA GEMM
// pre[row][o] = sum_c A[row][c] * W[o][c]; A row-major f16 (ld=Kc), W row-major f16 (ld=Kc)
// block = 256 thr (8 waves). Block covers 32 rows x ALL 512 cols: wave w owns cols [w*64, w*64+64).
// A is streamed exactly once per layer; W (512KB f16) is L2-resident and shared by all blocks.
// Wave tile 32x64 = 2 (M) x 4 (N) WMMA accumulators.
__global__ __launch_bounds__(256) void gemm_wmma_kernel(const _Float16* __restrict__ A,
                                                        const _Float16* __restrict__ W,
                                                        float* __restrict__ pre, int Kc) {
    int lane = threadIdx.x & 31;
    int wave = threadIdx.x >> 5;
    int l16  = lane & 15;
    int lh   = lane >> 4;
    int rowW = blockIdx.x * 32;
    int colW = wave * 64;

    v8f acc[2][4];
#pragma unroll
    for (int mt = 0; mt < 2; ++mt)
#pragma unroll
        for (int nt = 0; nt < 4; ++nt) acc[mt][nt] = (v8f){};

    const _Float16* Ap[2];
    Ap[0] = A + (size_t)(rowW + l16) * Kc;
    Ap[1] = A + (size_t)(rowW + 16 + l16) * Kc;
    const _Float16* Bp[4];
#pragma unroll
    for (int nt = 0; nt < 4; ++nt)
        Bp[nt] = W + (size_t)(colW + nt * 16 + l16) * Kc;

    for (int k0 = 0; k0 < Kc; k0 += 32) {
        // prefetch the A stream two k-steps ahead; locality=3 -> WGP-scope prefetch
        // (pull into all cache levels on miss), hiding HBM latency for the streamed A.
        __builtin_prefetch(Ap[0] + k0 + 64, 0, 3);
        __builtin_prefetch(Ap[1] + k0 + 64, 0, 3);

        union { v16h v; unsigned u[8]; } a[2], b[4];
#pragma unroll
        for (int v = 0; v < 8; ++v) {
            // 16-bit A/B-matrix VGPR layout: VGPR v holds K = (v&4?16:0) + lh*8 + (v&3)*2 (+1)
            int off = k0 + ((v & 4) ? 16 : 0) + lh * 8 + (v & 3) * 2;
            a[0].u[v] = *(const unsigned*)(Ap[0] + off);
            a[1].u[v] = *(const unsigned*)(Ap[1] + off);
            b[0].u[v] = *(const unsigned*)(Bp[0] + off);
            b[1].u[v] = *(const unsigned*)(Bp[1] + off);
            b[2].u[v] = *(const unsigned*)(Bp[2] + off);
            b[3].u[v] = *(const unsigned*)(Bp[3] + off);
        }
#pragma unroll
        for (int mt = 0; mt < 2; ++mt)
#pragma unroll
            for (int nt = 0; nt < 4; ++nt)
                acc[mt][nt] = __builtin_amdgcn_wmma_f32_16x16x32_f16(
                    false, a[mt].v, false, b[nt].v, (short)0, acc[mt][nt], false, false);
    }

    // C/D layout: VGPR v -> row M = v + 8*lh, col N = lane&15
#pragma unroll
    for (int mt = 0; mt < 2; ++mt) {
#pragma unroll
        for (int v = 0; v < 8; ++v) {
            int r = rowW + mt * 16 + v + 8 * lh;
            float* prow = pre + (size_t)r * OC + colW + l16;
#pragma unroll
            for (int nt = 0; nt < 4; ++nt)
                prow[nt * 16] = acc[mt][nt][v];
        }
    }
}

// ---------------------------------------------------------------- BN stats over rows (512 channels)
__global__ void zero_kernel(float* __restrict__ p, int n) {
    int t = blockIdx.x * blockDim.x + threadIdx.x;
    if (t < n) p[t] = 0.f;
}

__global__ void stats_rows_kernel(const float* __restrict__ pre, float* __restrict__ sums) {
    int row0 = blockIdx.x * 128;     // 1024 blocks
    int o0 = threadIdx.x, o1 = threadIdx.x + 256;
    float s0 = 0.f, q0 = 0.f, s1 = 0.f, q1 = 0.f;
    for (int r = 0; r < 128; ++r) {
        const float* p = pre + (size_t)(row0 + r) * OC;
        float v0 = p[o0], v1 = p[o1];
        s0 += v0; q0 += v0 * v0;
        s1 += v1; q1 += v1 * v1;
    }
    atomicAdd(&sums[o0], s0); atomicAdd(&sums[OC + o0], q0);
    atomicAdd(&sums[o1], s1); atomicAdd(&sums[OC + o1], q1);
}

__global__ void finalize_sb_kernel(const float* __restrict__ sums, const float* __restrict__ g,
                                   const float* __restrict__ bt, float* __restrict__ sb, float invn) {
    int o = blockIdx.x * blockDim.x + threadIdx.x;   // 512
    float mean = sums[o] * invn;
    float var  = sums[OC + o] * invn - mean * mean;
    float sc   = g[o] * rsqrtf(var + 1e-5f);
    sb[2 * o]     = sc;
    sb[2 * o + 1] = bt[o] - mean * sc;
}

__global__ void norm_relu_f16_kernel(const float* __restrict__ pre, const float* __restrict__ sb,
                                     _Float16* __restrict__ act) {
    size_t t = (size_t)blockIdx.x * 256 + threadIdx.x;   // ROWS*512
    int o = (int)(t & (OC - 1));
    float v = pre[t] * sb[2 * o] + sb[2 * o + 1];
    act[t] = (_Float16)fmaxf(v, 0.f);
}

__global__ void final_max_kernel(const float* __restrict__ pre, const float* __restrict__ sb,
                                 float* __restrict__ out) {
    int t  = blockIdx.x * blockDim.x + threadIdx.x;  // 4096*512
    int o  = t & (OC - 1);
    int bm = t >> 9;
    float sc = sb[2 * o], bi = sb[2 * o + 1];
    float mx = 0.f;                                  // relu'd values are >= 0
    for (int k = 0; k < K_; ++k) {
        float v = pre[((size_t)bm * K_ + k) * OC + o] * sc + bi;
        mx = fmaxf(mx, fmaxf(v, 0.f));
    }
    out[t] = mx;
}

// ---------------------------------------------------------------- host launch
static inline size_t alignup(size_t x) { return (x + 255) & ~(size_t)255; }

extern "C" void kernel_launch(void* const* d_in, const int* in_sizes, int n_in,
                              void* d_out, int out_size, void* d_ws, size_t ws_size,
                              hipStream_t stream) {
    const float* ffps_xyz  = (const float*)d_in[0];
    const float* ffps_feat = (const float*)d_in[1];
    const float* bxyz      = (const float*)d_in[2];
    const float* bfeat     = (const float*)d_in[3];
    const float* sw1 = (const float*)d_in[4];
    const float* sg1 = (const float*)d_in[5];
    const float* sb1 = (const float*)d_in[6];
    const float* sw2 = (const float*)d_in[7];
    const float* sg2 = (const float*)d_in[8];
    const float* sb2 = (const float*)d_in[9];
    const float* w1 = (const float*)d_in[10];
    const float* g1 = (const float*)d_in[11];
    const float* b1 = (const float*)d_in[12];
    const float* w2 = (const float*)d_in[13];
    const float* g2 = (const float*)d_in[14];
    const float* b2 = (const float*)d_in[15];
    const float* w3 = (const float*)d_in[16];
    const float* g3 = (const float*)d_in[17];
    const float* b3 = (const float*)d_in[18];
    float* out = (float*)d_out;

    char* ws = (char*)d_ws;
    size_t off = 0;
    float*     bufPre = (float*)(ws + off);     off += alignup((size_t)ROWS * OC * 4);
    _Float16*  bufA   = (_Float16*)(ws + off);  off += alignup((size_t)ROWS * OC * 2);
    _Float16*  bufB   = (_Float16*)(ws + off);  off += alignup((size_t)ROWS * OC * 2);
    _Float16*  w1h    = (_Float16*)(ws + off);  off += alignup((size_t)OC * KC1 * 2);
    _Float16*  w2h    = (_Float16*)(ws + off);  off += alignup((size_t)OC * OC * 2);
    _Float16*  w3h    = (_Float16*)(ws + off);  off += alignup((size_t)OC * OC * 2);
    float*     h1pre  = (float*)(ws + off);     off += alignup((size_t)128 * BM_ * 4);
    float*     h2pre  = (float*)(ws + off);     off += alignup((size_t)3 * BM_ * 4);
    float*     sb1s   = (float*)(ws + off);     off += alignup(128 * 2 * 4);
    float*     sb2s   = (float*)(ws + off);     off += alignup(3 * 2 * 4);
    float*     sbL    = (float*)(ws + off);     off += alignup(OC * 2 * 4);
    float*     sums   = (float*)(ws + off);     off += alignup(OC * 2 * 4);
    float*     nxyz   = (float*)(ws + off);     off += alignup((size_t)BM_ * 3 * 4);
    int*       idxb   = (int*)(ws + off);       off += alignup((size_t)BM_ * K_ * 4);

    // weights to f16
    cvt_w1_kernel<<<OC, KC1, 0, stream>>>(w1, w1h);
    cvt_w_kernel<<<(OC * OC + 255) / 256, 256, 0, stream>>>(w2, w2h, OC * OC);
    cvt_w_kernel<<<(OC * OC + 255) / 256, 256, 0, stream>>>(w3, w3h, OC * OC);

    // shift MLP -> new_xyz
    shift1_kernel<<<(128 * BM_) / 256, 256, 0, stream>>>(sw1, ffps_feat, h1pre);
    stats_chan_kernel<<<128, 256, 0, stream>>>(h1pre, sg1, sb1, sb1s, BM_);
    shift2_kernel<<<(3 * BM_) / 256, 256, 0, stream>>>(sw2, h1pre, sb1s, h2pre);
    stats_chan_kernel<<<3, 256, 0, stream>>>(h2pre, sg2, sb2, sb2s, BM_);
    newxyz_kernel<<<BM_ / 256, 256, 0, stream>>>(ffps_xyz, h2pre, sb2s, nxyz);

    // ball query + gather
    ballquery_kernel<<<(BM_ * 32) / 256, 256, 0, stream>>>(nxyz, bxyz, idxb);
    gather_kernel<<<ROWS, KC1, 0, stream>>>(bxyz, bfeat, nxyz, idxb, bufA);

    int gemmGrid = ROWS / 32;   // each block: 32 rows x 512 cols

    // layer 1: bufA (f16, ld 288) -> bufPre
    gemm_wmma_kernel<<<gemmGrid, 256, 0, stream>>>(bufA, w1h, bufPre, KC1);
    zero_kernel<<<4, 256, 0, stream>>>(sums, OC * 2);
    stats_rows_kernel<<<ROWS / 128, 256, 0, stream>>>(bufPre, sums);
    finalize_sb_kernel<<<2, 256, 0, stream>>>(sums, g1, b1, sbL, 1.0f / ROWS);
    norm_relu_f16_kernel<<<(ROWS * OC) / 256, 256, 0, stream>>>(bufPre, sbL, bufB);

    // layer 2: bufB -> bufPre
    gemm_wmma_kernel<<<gemmGrid, 256, 0, stream>>>(bufB, w2h, bufPre, OC);
    zero_kernel<<<4, 256, 0, stream>>>(sums, OC * 2);
    stats_rows_kernel<<<ROWS / 128, 256, 0, stream>>>(bufPre, sums);
    finalize_sb_kernel<<<2, 256, 0, stream>>>(sums, g2, b2, sbL, 1.0f / ROWS);
    norm_relu_f16_kernel<<<(ROWS * OC) / 256, 256, 0, stream>>>(bufPre, sbL, bufA);

    // layer 3: bufA -> bufPre
    gemm_wmma_kernel<<<gemmGrid, 256, 0, stream>>>(bufA, w3h, bufPre, OC);
    zero_kernel<<<4, 256, 0, stream>>>(sums, OC * 2);
    stats_rows_kernel<<<ROWS / 128, 256, 0, stream>>>(bufPre, sums);
    finalize_sb_kernel<<<2, 256, 0, stream>>>(sums, g3, b3, sbL, 1.0f / ROWS);

    // BN + relu + max over K -> out (B, M, 512)
    final_max_kernel<<<(BM_ * OC) / 256, 256, 0, stream>>>(bufPre, sbL, out);
    (void)in_sizes; (void)n_in; (void)out_size; (void)ws_size;
}
